// GCN_EN_32169305047252
// MI455X (gfx1250) — matrix operand order repaired and checked by reading, not compiled, not run on hardware
//
#include <hip/hip_runtime.h>
#include <hip/hip_bf16.h>

#define NN 100000
#define NE 1600000
#define NG 4096

typedef __attribute__((ext_vector_type(16))) _Float16 v16h;
typedef __attribute__((ext_vector_type(2)))  _Float16 h2;
typedef __attribute__((ext_vector_type(8)))  float    v8f;

__device__ __forceinline__ float silu_f(float v) {
  // v * 1/(1+e^-v) with hardware rcp (v_exp_f32 + v_rcp_f32)
  return v * __builtin_amdgcn_rcpf(1.0f + __expf(-v));
}

// ---------------------------------------------------------------------------
// WMMA GEMM: C[M x Nc] = act(A[M x K] @ B[K x Nc] + bias)
// K, MODE compile-time: MODE 0 = raw, 1 = +bias, 2 = +bias+SiLU.
// USE_LDS: stage B panel in LDS as packed f16 pairs (needs Nc*K*2 bytes).
// M % 16 == 0, K % 32 == 0, Nc % 16 == 0. 256 threads = 8 waves; one 16-row
// strip per block; waves stride over N tiles.
// ---------------------------------------------------------------------------
template <int K, bool USE_LDS, int MODE>
__global__ __launch_bounds__(256) void k_gemm(
    const float* __restrict__ A, const float* __restrict__ B,
    const float* __restrict__ bias, float* __restrict__ C,
    int M, int Nc, int ldc)
{
  extern __shared__ unsigned sB[];              // [Nc][K/2] packed f16 pairs
  constexpr int KH = K / 2;

  if (USE_LDS) {
    // cooperative, coalesced fill: idx = kp*Nc + n
    for (int idx = threadIdx.x; idx < Nc * KH; idx += blockDim.x) {
      int n = idx % Nc, kp = idx / Nc;
      union { h2 h; unsigned u; } pk;
      pk.h[0] = (_Float16)B[(size_t)(2 * kp)     * Nc + n];
      pk.h[1] = (_Float16)B[(size_t)(2 * kp + 1) * Nc + n];
      sB[(size_t)n * KH + kp] = pk.u;
    }
    __syncthreads();
  }

  const int lane = threadIdx.x & 31;
  const int wave = threadIdx.x >> 5;
  const int row0 = blockIdx.x << 4;
  const int mlo  = lane & 15;
  const int hi   = (lane >> 4) & 1;             // lane half
  const int ntiles = Nc >> 4;

  for (int nt = wave; nt < ntiles; nt += 8) {
    const int n = (nt << 4) + mlo;
    v8f acc = {};
#pragma unroll
    for (int k0 = 0; k0 < K; k0 += 32) {
      // ---- A tile (16x32 f16): lane half h: contiguous K = 16h..16h+15
      v16h a;
      {
        const float* pa = A + (size_t)(row0 + mlo) * K + k0 + hi * 16;
        float4 q0 = *(const float4*)(pa);
        float4 q1 = *(const float4*)(pa + 4);
        float4 q2 = *(const float4*)(pa + 8);
        float4 q3 = *(const float4*)(pa + 12);
        a[0]=(_Float16)q0.x;  a[1]=(_Float16)q0.y;  a[2]=(_Float16)q0.z;  a[3]=(_Float16)q0.w;
        a[4]=(_Float16)q1.x;  a[5]=(_Float16)q1.y;  a[6]=(_Float16)q1.z;  a[7]=(_Float16)q1.w;
        a[8]=(_Float16)q2.x;  a[9]=(_Float16)q2.y;  a[10]=(_Float16)q2.z; a[11]=(_Float16)q2.w;
        a[12]=(_Float16)q3.x; a[13]=(_Float16)q3.y; a[14]=(_Float16)q3.z; a[15]=(_Float16)q3.w;
      }
      // ---- B tile (32x16 f16): lane half h: contiguous K = 16h..16h+15
      v16h b;
      if (USE_LDS) {
        const unsigned* pb = sB + (size_t)n * KH + (k0 >> 1) + hi * 8;
#pragma unroll
        for (int j = 0; j < 8; ++j) {
          union { unsigned u; h2 h; } pk; pk.u = pb[j];
          b[2 * j]     = pk.h[0];
          b[2 * j + 1] = pk.h[1];
        }
      } else {
        const float* pb = B + (size_t)(k0 + hi * 16) * Nc + n;
#pragma unroll
        for (int j = 0; j < 16; ++j) b[j] = (_Float16)pb[(size_t)j * Nc];
      }
      acc = __builtin_amdgcn_wmma_f32_16x16x32_f16(
          false, a, false, b, (short)0, acc, false, false);
    }
    const float bv = (MODE >= 1) ? bias[n] : 0.0f;
    const int mbase = row0 + hi * 8;
#pragma unroll
    for (int r = 0; r < 8; ++r) {
      float v = acc[r] + bv;
      if (MODE == 2) v = silu_f(v);
      C[(size_t)(mbase + r) * ldc + n] = v;
    }
  }
}

// ---------------------------------------------------------------------------
// Small elementwise / graph kernels
// ---------------------------------------------------------------------------
__global__ void k_fill(float* __restrict__ p, float v, int n) {
  int i = blockIdx.x * blockDim.x + threadIdx.x;
  if (i < n) p[i] = v;
}

__global__ void k_deg(const int* __restrict__ dst, float* __restrict__ deg, int E) {
  int e = blockIdx.x * blockDim.x + threadIdx.x;
  if (e < E) atomicAdd(&deg[dst[e]], 1.0f);
}

__global__ void k_rsqrt_ip(float* __restrict__ p, int n) {
  int i = blockIdx.x * blockDim.x + threadIdx.x;
  if (i < n) p[i] = rsqrtf(p[i]);
}

// fo (N x 64) in place: L2-normalize each row, scale by atom_ens
__global__ void k_en_norm(float* __restrict__ fo, const float* __restrict__ ens) {
  int i = blockIdx.x * blockDim.x + threadIdx.x;
  if (i >= NN) return;
  float* row = fo + (size_t)i * 64;
  float s = 0.0f;
#pragma unroll 8
  for (int f = 0; f < 64; ++f) { float v = row[f]; s += v * v; }
  float scale = ens[i] / fmaxf(sqrtf(s), 1e-12f);
#pragma unroll 8
  for (int f = 0; f < 64; ++f) row[f] *= scale;
}

// out[i][f] = bias[f] + t[i][f] * dinv[i]^2   (self-loop term + bias), F = 128
__global__ void k_prop_init(const float* __restrict__ t, const float* __restrict__ dinv,
                            const float* __restrict__ bias, float* __restrict__ out) {
  size_t idx = (size_t)blockIdx.x * blockDim.x + threadIdx.x;
  if (idx >= (size_t)NN * 128) return;
  int i = (int)(idx >> 7), f = (int)(idx & 127);
  float d = dinv[i];
  out[idx] = bias[f] + t[idx] * d * d;
}

// per-edge scaled scatter-add: 32 lanes per edge, 4 features each (F = 128)
__global__ void k_prop_edges(const int* __restrict__ src, const int* __restrict__ dst,
                             const float* __restrict__ t, const float* __restrict__ dinv,
                             float* __restrict__ out) {
  int gid = blockIdx.x * blockDim.x + threadIdx.x;
  int e = gid >> 5, lane = gid & 31;
  if (e >= NE) return;
  int s = src[e], d = dst[e];
  float nrm = dinv[s] * dinv[d];
  float4 v = *reinterpret_cast<const float4*>(t + (size_t)s * 128 + lane * 4);
  float* o = out + (size_t)d * 128 + lane * 4;
  atomicAdd(o + 0, v.x * nrm);
  atomicAdd(o + 1, v.y * nrm);
  atomicAdd(o + 2, v.z * nrm);
  atomicAdd(o + 3, v.w * nrm);
}

// batch stats per column: one block per column
__global__ void k_bn_stats(const float* __restrict__ X, int M, int F,
                           float* __restrict__ mean, float* __restrict__ var) {
  int c = blockIdx.x;
  float s = 0.0f, s2 = 0.0f;
  for (int r = threadIdx.x; r < M; r += blockDim.x) {
    float v = X[(size_t)r * F + c];
    s += v; s2 += v * v;
  }
  __shared__ float sh[256], sh2[256];
  sh[threadIdx.x] = s; sh2[threadIdx.x] = s2;
  __syncthreads();
  for (int off = blockDim.x >> 1; off > 0; off >>= 1) {
    if ((int)threadIdx.x < off) {
      sh[threadIdx.x]  += sh[threadIdx.x + off];
      sh2[threadIdx.x] += sh2[threadIdx.x + off];
    }
    __syncthreads();
  }
  if (threadIdx.x == 0) {
    float mu = sh[0] / (float)M;
    mean[c] = mu;
    var[c]  = sh2[0] / (float)M - mu * mu;
  }
}

__global__ void k_bn_silu(float* __restrict__ X, const float* __restrict__ mean,
                          const float* __restrict__ var, const float* __restrict__ g,
                          const float* __restrict__ b, int M, int F) {
  size_t idx = (size_t)blockIdx.x * blockDim.x + threadIdx.x;
  if (idx >= (size_t)M * F) return;
  int f = (int)(idx % F);
  float v = (X[idx] - mean[f]) * rsqrtf(var[f] + 1e-5f) * g[f] + b[f];
  X[idx] = silu_f(v);
}

__global__ void k_silu(float* __restrict__ X, size_t n) {
  size_t i = (size_t)blockIdx.x * blockDim.x + threadIdx.x;
  if (i < n) X[i] = silu_f(X[i]);
}

// global_add_pool into left half of fused (G x 256)
__global__ void k_pool(const float* __restrict__ h, const int* __restrict__ batch,
                       float* __restrict__ fused) {
  int gid = blockIdx.x * blockDim.x + threadIdx.x;
  int i = gid >> 5, lane = gid & 31;
  if (i >= NN) return;
  int g = batch[i];
  float4 v = *reinterpret_cast<const float4*>(h + (size_t)i * 128 + lane * 4);
  float* o = fused + (size_t)g * 256 + lane * 4;
  atomicAdd(o + 0, v.x);
  atomicAdd(o + 1, v.y);
  atomicAdd(o + 2, v.z);
  atomicAdd(o + 3, v.w);
}

// final 64 -> 1 projection
__global__ void k_head_final(const float* __restrict__ h, const float* __restrict__ w,
                             const float* __restrict__ b, float* __restrict__ out) {
  int g = blockIdx.x * blockDim.x + threadIdx.x;
  if (g >= NG) return;
  const float* row = h + (size_t)g * 64;
  float s = b[0];
#pragma unroll 8
  for (int f = 0; f < 64; ++f) s += row[f] * w[f];
  out[g] = s;
}

// ---------------------------------------------------------------------------
extern "C" void kernel_launch(void* const* d_in, const int* in_sizes, int n_in,
                              void* d_out, int out_size, void* d_ws, size_t ws_size,
                              hipStream_t stream) {
  const float* x         = (const float*)d_in[0];
  const float* atom_ens  = (const float*)d_in[1];
  const float* mol_feats = (const float*)d_in[2];
  const int*   eidx      = (const int*)d_in[3];
  const int*   batch     = (const int*)d_in[4];
  const float* W_en = (const float*)d_in[5];  const float* b_en = (const float*)d_in[6];
  const float* gW0 = (const float*)d_in[7];   const float* gb0 = (const float*)d_in[8];
  const float* gW1 = (const float*)d_in[9];   const float* gb1 = (const float*)d_in[10];
  const float* gW2 = (const float*)d_in[11];  const float* gb2 = (const float*)d_in[12];
  const float* bn_gc_g = (const float*)d_in[13]; const float* bn_gc_b = (const float*)d_in[14];
  const float* mW0 = (const float*)d_in[15];  const float* mb0 = (const float*)d_in[16];
  const float* mW1 = (const float*)d_in[17];  const float* mb1 = (const float*)d_in[18];
  const float* mW2 = (const float*)d_in[19];  const float* mb2 = (const float*)d_in[20];
  const float* bn_m_g = (const float*)d_in[21]; const float* bn_m_b = (const float*)d_in[22];
  const float* fW0 = (const float*)d_in[23];  const float* fb0 = (const float*)d_in[24];
  const float* fW1 = (const float*)d_in[25];  const float* fb1 = (const float*)d_in[26];
  const float* fW2 = (const float*)d_in[27];  const float* fb2 = (const float*)d_in[28];
  const float* fW3 = (const float*)d_in[29];  const float* fb3 = (const float*)d_in[30];

  const int* src = eidx;          // edge_index[0]
  const int* dst = eidx + NE;     // edge_index[1]

  // workspace layout (floats)
  float* ws    = (float*)d_ws;
  float* dinv  = ws;                                // N
  float* en    = dinv + NN;                         // N*64
  float* bufA  = en + (size_t)NN * 64;              // N*128
  float* bufB  = bufA + (size_t)NN * 128;           // N*128
  float* fused = bufB + (size_t)NN * 128;           // G*256
  float* headA = fused + (size_t)NG * 256;          // G*256
  float* headB = headA + (size_t)NG * 256;          // G*256
  float* mol0  = headB + (size_t)NG * 256;          // G*128
  float* mol1  = mol0 + (size_t)NG * 128;           // G*128
  float* stats = mol1 + (size_t)NG * 128;           // 256 (mean|var)

  const int T = 256;
  auto blk = [](size_t n) { return (unsigned)((n + 255) / 256); };

  // LDS panel sizes (bytes) = Nc * K * 2
  constexpr size_t LDS_64x64   = 64  * 64  * 2;   //  8 KB
  constexpr size_t LDS_64x128  = 128 * 64  * 2;   // 16 KB
  constexpr size_t LDS_128x128 = 128 * 128 * 2;   // 32 KB
  constexpr size_t LDS_256x64  = 64  * 256 * 2;   // 32 KB

  // --- degree / normalization (self loops -> init to 1) ---
  k_fill<<<blk(NN), T, 0, stream>>>(dinv, 1.0f, NN);
  k_deg<<<blk(NE), T, 0, stream>>>(dst, dinv, NE);
  k_rsqrt_ip<<<blk(NN), T, 0, stream>>>(dinv, NN);

  // --- calc_en: linear -> row L2 norm -> scale ---
  k_gemm<64, true, 1><<<NN / 16, T, LDS_64x64, stream>>>(x, W_en, b_en, en, NN, 64, 64);
  k_en_norm<<<blk(NN), T, 0, stream>>>(en, atom_ens);

  // --- GCN layer 0: transform, propagate, BN, SiLU ---
  k_gemm<64, true, 0><<<NN / 16, T, LDS_64x128, stream>>>(en, gW0, nullptr, bufB, NN, 128, 128);
  k_prop_init<<<blk((size_t)NN * 128), T, 0, stream>>>(bufB, dinv, gb0, bufA);
  k_prop_edges<<<blk((size_t)NE * 32), T, 0, stream>>>(src, dst, bufB, dinv, bufA);
  k_bn_stats<<<128, T, 0, stream>>>(bufA, NN, 128, stats, stats + 128);
  k_bn_silu<<<blk((size_t)NN * 128), T, 0, stream>>>(bufA, stats, stats + 128,
                                                     bn_gc_g, bn_gc_b, NN, 128);

  // --- GCN layer 1 ---
  k_gemm<128, true, 0><<<NN / 16, T, LDS_128x128, stream>>>(bufA, gW1, nullptr, bufB, NN, 128, 128);
  k_prop_init<<<blk((size_t)NN * 128), T, 0, stream>>>(bufB, dinv, gb1, bufA);
  k_prop_edges<<<blk((size_t)NE * 32), T, 0, stream>>>(src, dst, bufB, dinv, bufA);
  k_silu<<<blk((size_t)NN * 128), T, 0, stream>>>(bufA, (size_t)NN * 128);

  // --- GCN layer 2 ---
  k_gemm<128, true, 0><<<NN / 16, T, LDS_128x128, stream>>>(bufA, gW2, nullptr, bufB, NN, 128, 128);
  k_prop_init<<<blk((size_t)NN * 128), T, 0, stream>>>(bufB, dinv, gb2, bufA);
  k_prop_edges<<<blk((size_t)NE * 32), T, 0, stream>>>(src, dst, bufB, dinv, bufA);
  k_silu<<<blk((size_t)NN * 128), T, 0, stream>>>(bufA, (size_t)NN * 128);

  // --- global_add_pool into fused[:, 0:128] ---
  k_fill<<<blk((size_t)NG * 256), T, 0, stream>>>(fused, 0.0f, NG * 256);
  k_pool<<<blk((size_t)NN * 32), T, 0, stream>>>(bufA, batch, fused);

  // --- mol branch ---
  k_gemm<128, true, 1><<<NG / 16, T, LDS_128x128, stream>>>(mol_feats, mW0, mb0, mol0, NG, 128, 128);
  k_bn_stats<<<128, T, 0, stream>>>(mol0, NG, 128, stats, stats + 128);
  k_bn_silu<<<blk((size_t)NG * 128), T, 0, stream>>>(mol0, stats, stats + 128,
                                                     bn_m_g, bn_m_b, NG, 128);
  k_gemm<128, true, 2><<<NG / 16, T, LDS_128x128, stream>>>(mol0, mW1, mb1, mol1, NG, 128, 128);
  // write directly into fused[:, 128:256] via ldc = 256
  k_gemm<128, true, 2><<<NG / 16, T, LDS_128x128, stream>>>(mol1, mW2, mb2, fused + 128, NG, 128, 256);

  // --- fused head (256x256 panels too big for default LDS cap: direct-global B) ---
  k_gemm<256, false, 2><<<NG / 16, T, 0, stream>>>(fused, fW0, fb0, headA, NG, 256, 256);
  k_gemm<256, false, 2><<<NG / 16, T, 0, stream>>>(headA, fW1, fb1, headB, NG, 256, 256);
  k_gemm<256, true, 2><<<NG / 16, T, LDS_256x64, stream>>>(headB, fW2, fb2, headA, NG, 64, 64);
  k_head_final<<<blk(NG), T, 0, stream>>>(headA, fW3, fb3, (float*)d_out);
}
